// PathwayGCN_2061584302287
// MI455X (gfx1250) — matrix operand order repaired and checked by reading, not compiled, not run on hardware
//
#include <hip/hip_runtime.h>

// ---------------------------------------------------------------------------
// Types for CDNA5 WMMA (wave32): v_wmma_f32_16x16x32_bf16
// ---------------------------------------------------------------------------
typedef __attribute__((ext_vector_type(16))) __bf16 v16bf;
typedef __attribute__((ext_vector_type(8)))  float  v8f;

#define TILE_K 32

__device__ __forceinline__ unsigned pk_bf16(float a, float b) {
    union { __bf16 h[2]; unsigned u; } p;
    p.h[0] = (__bf16)a;
    p.h[1] = (__bf16)b;
    return p.u;
}

// ---------------------------------------------------------------------------
// Degree / normalization kernels
// ---------------------------------------------------------------------------
__global__ void init_deg_kernel(float* __restrict__ deg, int N) {
    int i = blockIdx.x * blockDim.x + threadIdx.x;
    if (i < N) deg[i] = 1.0f;   // self-loop weight 1 folded into init
}

__global__ void deg_atomic_kernel(const int* __restrict__ dst,
                                  const float* __restrict__ ew,
                                  float* __restrict__ deg, int E) {
    int e = blockIdx.x * blockDim.x + threadIdx.x;
    if (e < E) atomicAdd(&deg[dst[e]], ew[e]);
}

__global__ void dinv_kernel(float* __restrict__ deg, int N) {
    int i = blockIdx.x * blockDim.x + threadIdx.x;
    if (i < N) {
        float d = deg[i];
        deg[i] = (d > 0.0f) ? (1.0f / sqrtf(d)) : 0.0f;   // in-place -> dinv
    }
}

__global__ void norm_kernel(const int* __restrict__ src, const int* __restrict__ dst,
                            const float* __restrict__ ew, const float* __restrict__ dinv,
                            float* __restrict__ norm, int E) {
    int e = blockIdx.x * blockDim.x + threadIdx.x;
    if (e < E) norm[e] = dinv[src[e]] * ew[e] * dinv[dst[e]];
}

// ---------------------------------------------------------------------------
// WMMA GEMM: C[N x F] = A[N x K] * B[K x F]   (f32 in, bf16 WMMA, f32 accum)
// Block = 128 threads (4 waves). Block tile: 64 rows x 64 cols.
// Wave w owns rows [w*16, w*16+16) x all 64 cols -> 4 accumulators,
// 4 WMMAs per K-chunk reusing one A fragment from registers.
// Main K-loop runs only full 32-chunks with branch-free staging
// (clamped-index loads + selects); the K tail is a single extra chunk.
// Global loads for chunk k+1 are issued before the WMMAs of chunk k.
// ---------------------------------------------------------------------------
__global__ __launch_bounds__(128) void gemm_bf16_wmma_kernel(
    const float* __restrict__ A, const float* __restrict__ B,
    float* __restrict__ C, int N, int K, int F) {
    __shared__ __bf16 sA[64][TILE_K];   // 64 rows x 32 K   (4 KB), row-major
    __shared__ __bf16 sBt[64][TILE_K];  // 64 cols x 32 K   (4 KB), col-major

    const int tid  = threadIdx.x;
    const int wave = tid >> 5;
    const int lane = tid & 31;
    const int row0 = blockIdx.y * 64;
    const int colB = blockIdx.x * 64;

    const int half = lane >> 4;   // 0: lanes 0-15, 1: lanes 16-31
    const int mr   = lane & 15;

    v8f acc[4] = {};
    union Frag { v16bf v; uint4 q[2]; };

    const int Kfull = K & ~(TILE_K - 1);

    float2 va[8];       // A stage regs: 8 float2 segments per thread
    float  vb[4][4];    // B stage regs: 4 (4k x 1n) segments per thread

    // ---- staging helpers (branch-free: clamped loads + selects) ----
    auto loadA_full = [&](int kk) {
        #pragma unroll
        for (int i = 0; i < 8; ++i) {
            int s = tid + i * 128;                 // 1024 float2 segments
            int m = s >> 4;
            int k = (s & 15) * 2;
            int gr  = row0 + m;
            int grc = (gr < N) ? gr : (N - 1);
            float2 v = *reinterpret_cast<const float2*>(A + (size_t)grc * K + kk + k);
            va[i].x = (gr < N) ? v.x : 0.0f;
            va[i].y = (gr < N) ? v.y : 0.0f;
        }
    };
    auto loadA_tail = [&](int kk) {
        #pragma unroll
        for (int i = 0; i < 8; ++i) {
            int s = tid + i * 128;
            int m = s >> 4;
            int k = (s & 15) * 2;
            int gr  = row0 + m;
            int grc = (gr < N) ? gr : (N - 1);
            int g0 = kk + k, g1 = kk + k + 1;
            int c0 = (g0 < K) ? g0 : (K - 1);
            int c1 = (g1 < K) ? g1 : (K - 1);
            float x = A[(size_t)grc * K + c0];
            float y = A[(size_t)grc * K + c1];
            va[i].x = (gr < N && g0 < K) ? x : 0.0f;
            va[i].y = (gr < N && g1 < K) ? y : 0.0f;
        }
    };
    auto loadB_full = [&](int kk) {
        #pragma unroll
        for (int i = 0; i < 4; ++i) {
            int s = tid + i * 128;                 // 512 segments
            int n = s & 63;
            int k4 = (s >> 6) * 4;
            #pragma unroll
            for (int j = 0; j < 4; ++j)
                vb[i][j] = B[(size_t)(kk + k4 + j) * F + colB + n];
        }
    };
    auto loadB_tail = [&](int kk) {
        #pragma unroll
        for (int i = 0; i < 4; ++i) {
            int s = tid + i * 128;
            int n = s & 63;
            int k4 = (s >> 6) * 4;
            #pragma unroll
            for (int j = 0; j < 4; ++j) {
                int gk = kk + k4 + j;
                int gc = (gk < K) ? gk : (K - 1);
                float v = B[(size_t)gc * F + colB + n];
                vb[i][j] = (gk < K) ? v : 0.0f;
            }
        }
    };
    auto store_stage = [&]() {
        #pragma unroll
        for (int i = 0; i < 8; ++i) {
            int s = tid + i * 128;
            int m = s >> 4;
            int k = (s & 15) * 2;
            *reinterpret_cast<unsigned*>(&sA[m][k]) = pk_bf16(va[i].x, va[i].y);
        }
        #pragma unroll
        for (int i = 0; i < 4; ++i) {
            int s = tid + i * 128;
            int n = s & 63;
            int k4 = (s >> 6) * 4;
            uint2 p;
            p.x = pk_bf16(vb[i][0], vb[i][1]);
            p.y = pk_bf16(vb[i][2], vb[i][3]);
            *reinterpret_cast<uint2*>(&sBt[n][k4]) = p;
        }
    };
    auto compute_chunk = [&]() {
        Frag a;
        const __bf16* aRow = &sA[wave * 16 + mr][0];
        a.q[0] = *reinterpret_cast<const uint4*>(aRow + half * 8);        // K = half*8 + 0..7
        a.q[1] = *reinterpret_cast<const uint4*>(aRow + 16 + half * 8);   // K = 16 + half*8 + 0..7
        #pragma unroll
        for (int c = 0; c < 4; ++c) {
            Frag b;
            const __bf16* bCol = &sBt[c * 16 + mr][0];
            b.q[0] = *reinterpret_cast<const uint4*>(bCol + half * 8);       // K = half*8 + 0..7
            b.q[1] = *reinterpret_cast<const uint4*>(bCol + 16 + half * 8);  // K = 16 + half*8 + 0..7
            acc[c] = __builtin_amdgcn_wmma_f32_16x16x32_bf16(
                /*neg_a=*/false, a.v, /*neg_b=*/false, b.v,
                /*c_mod=*/(short)0, acc[c], /*reuse_a=*/false, /*reuse_b=*/false);
        }
    };

    // ---- pipelined main loop over full chunks + one tail chunk ----
    if (Kfull > 0)      { loadA_full(0); loadB_full(0); }
    else if (K > 0)     { loadA_tail(0); loadB_tail(0); }

    for (int kk = 0; kk < Kfull; kk += TILE_K) {
        store_stage();
        __syncthreads();
        int kn = kk + TILE_K;
        if (kn < Kfull)      { loadA_full(kn); loadB_full(kn); }  // prefetch next
        else if (Kfull < K)  { loadA_tail(Kfull); loadB_tail(Kfull); }
        compute_chunk();
        __syncthreads();
    }
    if (Kfull < K) {
        store_stage();
        __syncthreads();
        compute_chunk();
    }

    // D layout: lane -> (n = lane%16), VGPR r -> M = r + 8*(lane/16)
    #pragma unroll
    for (int c = 0; c < 4; ++c) {
        const int col = colB + c * 16 + mr;
        #pragma unroll
        for (int r = 0; r < 8; ++r) {
            int gr = row0 + wave * 16 + r + 8 * half;
            if (gr < N) C[(size_t)gr * F + col] = acc[c][r];
        }
    }
}

// ---------------------------------------------------------------------------
// Edge scatter: out[dst] += norm[e] * X[src]   (L2-resident gather, f32 atomics)
// One thread per (edge, 4 consecutive features) -> float4 gather, 4 atomics.
// ---------------------------------------------------------------------------
__global__ __launch_bounds__(256) void scatter_edges_kernel(
    const int* __restrict__ src, const int* __restrict__ dst,
    const float* __restrict__ norm, const float* __restrict__ X,
    float* __restrict__ out, int E, int F) {
    long long idx = (long long)blockIdx.x * blockDim.x + threadIdx.x;
    int per_edge = F >> 2;
    long long total = (long long)E * per_edge;
    if (idx >= total) return;
    int e = (int)(idx / per_edge);
    int f = (int)(idx % per_edge) * 4;
    float w = norm[e];
    int s = src[e], d = dst[e];
    const float4 v = *reinterpret_cast<const float4*>(X + (size_t)s * F + f);
    float* o = out + (size_t)d * F + f;
    atomicAdd(o + 0, w * v.x);
    atomicAdd(o + 1, w * v.y);
    atomicAdd(o + 2, w * v.z);
    atomicAdd(o + 3, w * v.w);
}

// ---------------------------------------------------------------------------
// Finalize: H = (relu?)(H + dinv[i]^2 * XW + bias[f])   (self-loop + bias fused)
// ---------------------------------------------------------------------------
__global__ __launch_bounds__(256) void finalize_kernel(
    float* __restrict__ H, const float* __restrict__ XW,
    const float* __restrict__ dinv, const float* __restrict__ bias,
    int N, int F, int do_relu) {
    long long idx = (long long)blockIdx.x * blockDim.x + threadIdx.x;
    long long total = (long long)N * F;
    if (idx >= total) return;
    int i = (int)(idx / F);
    int f = (int)(idx % F);
    float di = dinv[i];
    float v = H[idx] + di * di * XW[idx] + bias[f];
    H[idx] = do_relu ? fmaxf(v, 0.0f) : v;
}

// ---------------------------------------------------------------------------
// Host-side launch
// ---------------------------------------------------------------------------
extern "C" void kernel_launch(void* const* d_in, const int* in_sizes, int n_in,
                              void* d_out, int out_size, void* d_ws, size_t ws_size,
                              hipStream_t stream) {
    const float* x  = (const float*)d_in[0];
    const int*   ei = (const int*)d_in[1];
    const float* ew = (const float*)d_in[2];
    const float* W1 = (const float*)d_in[3];
    const float* b1 = (const float*)d_in[4];
    const float* W2 = (const float*)d_in[5];
    const float* b2 = (const float*)d_in[6];

    const int F_in = 394, H = 256, F_out = 64;
    const int N = in_sizes[0] / F_in;
    const int E = in_sizes[2];
    const int* srcI = ei;       // edge_index[0]
    const int* dstI = ei + E;   // edge_index[1]

    // Workspace layout (floats): dinv[N] | norm[E] | xw[N*H] (reused as hw) | h[N*H]
    float* dinv = (float*)d_ws;
    float* norm = dinv + N;
    float* xw   = norm + E;
    float* h    = xw + (size_t)N * H;
    float* hw   = xw;                 // xw dead after layer-1 finalize -> reuse
    float* out  = (float*)d_out;

    // --- degree + symmetric normalization -------------------------------
    init_deg_kernel<<<(N + 255) / 256, 256, 0, stream>>>(dinv, N);
    deg_atomic_kernel<<<(E + 255) / 256, 256, 0, stream>>>(dstI, ew, dinv, E);
    dinv_kernel<<<(N + 255) / 256, 256, 0, stream>>>(dinv, N);
    norm_kernel<<<(E + 255) / 256, 256, 0, stream>>>(srcI, dstI, ew, dinv, norm, E);

    // --- layer 1: xw = x @ W1 (WMMA), aggregate, +self-loop +b1, ReLU ----
    hipMemsetAsync(h, 0, (size_t)N * H * sizeof(float), stream);
    {
        dim3 grid(H / 64, (N + 63) / 64);
        gemm_bf16_wmma_kernel<<<grid, 128, 0, stream>>>(x, W1, xw, N, F_in, H);
    }
    {
        long long total = (long long)E * (H / 4);
        int blocks = (int)((total + 255) / 256);
        scatter_edges_kernel<<<blocks, 256, 0, stream>>>(srcI, dstI, norm, xw, h, E, H);
    }
    {
        long long total = (long long)N * H;
        int blocks = (int)((total + 255) / 256);
        finalize_kernel<<<blocks, 256, 0, stream>>>(h, xw, dinv, b1, N, H, /*relu=*/1);
    }

    // --- layer 2: hw = h @ W2 (WMMA), aggregate into d_out, +self +b2 ----
    {
        dim3 grid(F_out / 64, (N + 63) / 64);
        gemm_bf16_wmma_kernel<<<grid, 128, 0, stream>>>(h, W2, hw, N, H, F_out);
    }
    hipMemsetAsync(out, 0, (size_t)N * F_out * sizeof(float), stream);
    {
        long long total = (long long)E * (F_out / 4);
        int blocks = (int)((total + 255) / 256);
        scatter_edges_kernel<<<blocks, 256, 0, stream>>>(srcI, dstI, norm, hw, out, E, F_out);
    }
    {
        long long total = (long long)N * F_out;
        int blocks = (int)((total + 255) / 256);
        finalize_kernel<<<blocks, 256, 0, stream>>>(out, hw, dinv, b2, N, F_out, /*relu=*/0);
    }
}
